// VardropLSTMStack_15685220565076
// MI455X (gfx1250) — matrix operand order repaired and checked by reading, not compiled, not run on hardware
//
#include <hip/hip_runtime.h>
#include <hip/hip_bf16.h>
#include <math.h>

// ---------------------------------------------------------------------------
// VardropLSTMStack for MI455X (gfx1250, wave32, WMMA bf16->f32)
// ---------------------------------------------------------------------------
// Shapes: B=32, S=256, D_IN=512, D=1024, H=512, 4H=2048, both dirs NC=4096.
//   * fp32 -> bf16 operands, fp32 accumulate (v_wmma_f32_16x16x32_bf16)
//   * weights & activations pre-swizzled into exact WMMA fragment layouts
//   * recurrence distributed over 8 WGs/direction; h double-buffered in
//     global (L2), c held in WMMA accumulator registers, per-step
//     device-scope atomic barrier; XG/residual register-prefetched one
//     step ahead so their L2 latency hides under the WMMA k-loop
// ---------------------------------------------------------------------------

typedef __attribute__((ext_vector_type(16))) __bf16 v16bf;
typedef __attribute__((ext_vector_type(8)))  float  v8f;

#define BB    32
#define SS    256
#define DIN   512
#define DD    1024
#define HH    512
#define G4    2048
#define NC    4096          // gates of both directions, dir-major
#define MROWS (BB*SS)       // 8192
#define NLAYER 4
#define NWGD  8             // workgroups per direction in the recurrence

union FragU { v16bf v; uint4 q[2]; };

__device__ __forceinline__ float sig_fast(float x) {
  return 1.0f / (1.0f + __expf(-x));
}
__device__ __forceinline__ float tanh_fast(float x) {
  return 2.0f / (1.0f + __expf(-2.0f * x)) - 1.0f;
}

// ---------------------------------------------------------------------------
// Pack fp32 row-major A (M x K) into WMMA 16x32 A-fragments (bf16).
// lane l: g = l>>4, m = l&15; elem e<8 -> k = 8g+e ; elem e>=8 -> k = 8+8g+e
// ---------------------------------------------------------------------------
__global__ __launch_bounds__(256) void pack_a_bf16(
    const float* __restrict__ A, __bf16* __restrict__ out, int Mt, int KF, int K) {
  int idx = blockIdx.x * blockDim.x + threadIdx.x;
  int total = Mt * KF * 32;
  if (idx >= total) return;
  int lane = idx & 31;
  int frag = idx >> 5;
  int kf   = frag % KF;
  int mt   = frag / KF;
  int g    = lane >> 4;
  int m    = mt * 16 + (lane & 15);
  const float* row = A + (size_t)m * K + kf * 32 + 8 * g;
  __bf16* o = out + (size_t)idx * 16;
#pragma unroll
  for (int e = 0; e < 8; ++e) o[e]     = (__bf16)row[e];
#pragma unroll
  for (int e = 0; e < 8; ++e) o[8 + e] = (__bf16)row[16 + e];
}

// ---------------------------------------------------------------------------
// Pack fp32 row-major W (N x K) into WMMA 32x16 B-fragments (bf16).
// lane l: g = l>>4, n = l&15; elem e<8 -> k = 8g+e ; elem e>=8 -> k = 8+8g+e
// ---------------------------------------------------------------------------
__global__ __launch_bounds__(256) void pack_b_bf16(
    const float* __restrict__ W, __bf16* __restrict__ out, int Nt, int KF, int K) {
  int idx = blockIdx.x * blockDim.x + threadIdx.x;
  int total = Nt * KF * 32;
  if (idx >= total) return;
  int lane = idx & 31;
  int frag = idx >> 5;
  int kf   = frag % KF;
  int nt   = frag / KF;
  int g    = lane >> 4;
  int n    = nt * 16 + (lane & 15);
  const float* row = W + (size_t)n * K + kf * 32 + 8 * g;
  __bf16* o = out + (size_t)idx * 16;
#pragma unroll
  for (int e = 0; e < 8; ++e) o[e]     = (__bf16)row[e];
#pragma unroll
  for (int e = 0; e < 8; ++e) o[8 + e] = (__bf16)row[16 + e];
}

// ---------------------------------------------------------------------------
// C(M,N) = Apack(M,K) * Bpack(N,K)^T [+ bias1[n] + bias2[n]], fp32 out.
// Block tile 128x128, 8 waves of (64 M x 32 N), 8 accumulators per wave.
// ---------------------------------------------------------------------------
__global__ __launch_bounds__(256) void gemm_wmma_bf16(
    const v16bf* __restrict__ Ap, const v16bf* __restrict__ Bp,
    float* __restrict__ Cout, int N, int KF,
    const float* __restrict__ bias1, const float* __restrict__ bias2) {
  const int w    = threadIdx.x >> 5;
  const int lane = threadIdx.x & 31;
  const int wm = w >> 2;          // 0..1 -> 4 mtiles each
  const int wn = w & 3;           // 0..3 -> 2 ntiles each
  const int mt0 = blockIdx.y * 8 + wm * 4;
  const int nt0 = blockIdx.x * 8 + wn * 2;

  v8f acc[4][2];
#pragma unroll
  for (int i = 0; i < 4; ++i)
#pragma unroll
    for (int j = 0; j < 2; ++j) acc[i][j] = {};

  for (int kf = 0; kf < KF; ++kf) {
    v16bf a[4], b[2];
#pragma unroll
    for (int i = 0; i < 4; ++i)
      a[i] = Ap[((size_t)(mt0 + i) * KF + kf) * 32 + lane];
#pragma unroll
    for (int j = 0; j < 2; ++j)
      b[j] = Bp[((size_t)(nt0 + j) * KF + kf) * 32 + lane];
#pragma unroll
    for (int i = 0; i < 4; ++i)
#pragma unroll
      for (int j = 0; j < 2; ++j)
        acc[i][j] = __builtin_amdgcn_wmma_f32_16x16x32_bf16(
            false, a[i], false, b[j], (short)0, acc[i][j], false, false);
  }

  const int g = lane >> 4, nl = lane & 15;
#pragma unroll
  for (int i = 0; i < 4; ++i) {
#pragma unroll
    for (int j = 0; j < 2; ++j) {
      int n = (nt0 + j) * 16 + nl;
      float badd = bias1 ? (bias1[n] + bias2[n]) : 0.0f;
#pragma unroll
      for (int r = 0; r < 8; ++r) {
        int m = (mt0 + i) * 16 + 8 * g + r;
        Cout[(size_t)m * N + n] = acc[i][j][r] + badd;
      }
    }
  }
}

// ---------------------------------------------------------------------------
// Distributed sequential bidirectional LSTM layer.
// grid.x = 2*NWGD (dir-major), block = 128 (4 waves).
// Global wave gw = wg*4 + w owns hidden slice j in [16gw, 16gw+16) and gate
// n-tiles q*32+gw (q = i,f,g,o), so cell state c lives in that wave's WMMA
// accumulator registers for all 256 steps.
// h is double-buffered in global memory (L2-resident, 32KB/dir/buf); one
// device-scope atomic barrier per step per direction orders buffer flips.
// XG gate pre-activations and the residual input are register-prefetched one
// timestep ahead so their L2 latency overlaps the WMMA k-loop instead of
// sitting on the serial dependence chain. Residual add fused into the store.
// ---------------------------------------------------------------------------
__global__ __launch_bounds__(128) void lstm_layer_recur(
    const v16bf* __restrict__ Whhp,    // packed (NC/16 ntiles)(16 kf)(32 lanes)
    const float* __restrict__ XG,      // (MROWS, NC) : x-gates + biases
    const float* __restrict__ Yprev,   // (MROWS, DD) : residual input
    float* __restrict__ Ynext,         // (MROWS, DD)
    __bf16* __restrict__ hglob,        // [2 dir][2 buf][32][512] bf16 (zeroed)
    unsigned int* __restrict__ ctr) {  // [2] step barrier counters (zeroed)
  const int dir  = blockIdx.x / NWGD;
  const int wg   = blockIdx.x % NWGD;
  const int tid  = threadIdx.x;
  const int w    = tid >> 5;           // 0..3
  const int lane = tid & 31;
  const int g    = lane >> 4;
  const int nl   = lane & 15;
  const int gw   = wg * 4 + w;         // 0..31 : global wave id in direction
  const int jbase = gw * 16;
  const int KF = HH / 32;              // 16
  const int j = jbase + nl;

  __bf16* hdir = hglob + (size_t)dir * 2 * BB * HH;

  float cst[2][8];
#pragma unroll
  for (int mt = 0; mt < 2; ++mt)
#pragma unroll
    for (int r = 0; r < 8; ++r) cst[mt][r] = 0.0f;

  // register double-buffers: x-gates and residual for the *current* step
  float xg[4][2][8];
  float ypf[2][8];
  {
    const int s0 = dir ? (SS - 1) : 0;
#pragma unroll
    for (int q = 0; q < 4; ++q) {
      const int col = dir * G4 + q * HH + j;
#pragma unroll
      for (int mt = 0; mt < 2; ++mt)
#pragma unroll
        for (int r = 0; r < 8; ++r) {
          int m = mt * 16 + 8 * g + r;
          xg[q][mt][r] = XG[(size_t)(m * SS + s0) * NC + col];
        }
    }
#pragma unroll
    for (int mt = 0; mt < 2; ++mt)
#pragma unroll
      for (int r = 0; r < 8; ++r) {
        int m = mt * 16 + 8 * g + r;
        ypf[mt][r] = Yprev[(size_t)(m * SS + s0) * DD + dir * HH + j];
      }
  }

  for (int t = 0; t < SS; ++t) {
    const int s = dir ? (SS - 1 - t) : t;
    const __bf16* hrd = hdir + (size_t)(t & 1) * BB * HH;        // read buffer
    __bf16*       hwr = hdir + (size_t)((t + 1) & 1) * BB * HH;  // write buffer

    // move prefetched x-gates into the accumulators
    v8f acc[4][2];
#pragma unroll
    for (int q = 0; q < 4; ++q)
#pragma unroll
      for (int mt = 0; mt < 2; ++mt)
#pragma unroll
        for (int r = 0; r < 8; ++r) acc[q][mt][r] = xg[q][mt][r];

    // issue next step's x-gate loads now; they complete under the k-loop
    if (t + 1 < SS) {
      const int sn = dir ? (SS - 2 - t) : (t + 1);
#pragma unroll
      for (int q = 0; q < 4; ++q) {
        const int col = dir * G4 + q * HH + j;
#pragma unroll
        for (int mt = 0; mt < 2; ++mt)
#pragma unroll
          for (int r = 0; r < 8; ++r) {
            int m = mt * 16 + 8 * g + r;
            xg[q][mt][r] = XG[(size_t)(m * SS + sn) * NC + col];
          }
      }
    }

    // gates += h @ Whh^T   (K = 512); A fragments gathered from global h
    for (int kf = 0; kf < KF; ++kf) {
      FragU a[2];
#pragma unroll
      for (int mt = 0; mt < 2; ++mt) {
        int mrow = mt * 16 + nl;
        const __bf16* hp = hrd + (size_t)mrow * HH + kf * 32 + 8 * g;
        a[mt].q[0] = *(const uint4*)hp;
        a[mt].q[1] = *(const uint4*)(hp + 16);
      }
#pragma unroll
      for (int q = 0; q < 4; ++q) {
        int ntile = dir * 128 + q * 32 + gw;
        v16bf b = Whhp[((size_t)ntile * KF + kf) * 32 + lane];
#pragma unroll
        for (int mt = 0; mt < 2; ++mt)
          acc[q][mt] = __builtin_amdgcn_wmma_f32_16x16x32_bf16(
              false, a[mt].v, false, b, (short)0, acc[q][mt], false, false);
      }
    }

    // elementwise gate nonlinearity, cell update, fused residual store
#pragma unroll
    for (int mt = 0; mt < 2; ++mt) {
#pragma unroll
      for (int r = 0; r < 8; ++r) {
        float iv = sig_fast (acc[0][mt][r]);
        float fv = sig_fast (acc[1][mt][r]);
        float gv = tanh_fast(acc[2][mt][r]);
        float ov = sig_fast (acc[3][mt][r]);
        float c  = fv * cst[mt][r] + iv * gv;
        cst[mt][r] = c;
        float h = ov * tanh_fast(c);
        int m = mt * 16 + 8 * g + r;
        size_t oidx = (size_t)(m * SS + s) * DD + dir * HH + j;
        Ynext[oidx] = h + ypf[mt][r];
        hwr[(size_t)m * HH + j] = (__bf16)h;
      }
    }

    // issue next step's residual loads (drain during barrier + next k-loop)
    if (t + 1 < SS) {
      const int sn = dir ? (SS - 2 - t) : (t + 1);
#pragma unroll
      for (int mt = 0; mt < 2; ++mt)
#pragma unroll
        for (int r = 0; r < 8; ++r) {
          int m = mt * 16 + 8 * g + r;
          ypf[mt][r] = Yprev[(size_t)(m * SS + sn) * DD + dir * HH + j];
        }
    }

    // per-direction grid barrier: all 8 WGs finish step t before any starts t+1
    if (t + 1 < SS) {
      __threadfence();             // make my h stores device-visible
      __syncthreads();
      if (tid == 0) {
        __hip_atomic_fetch_add(ctr + dir, 1u, __ATOMIC_RELEASE,
                               __HIP_MEMORY_SCOPE_AGENT);
        unsigned tgt = (unsigned)(t + 1) * NWGD;
        while (__hip_atomic_load(ctr + dir, __ATOMIC_ACQUIRE,
                                 __HIP_MEMORY_SCOPE_AGENT) < tgt)
          __builtin_amdgcn_s_sleep(1);
      }
      __syncthreads();
      __threadfence();             // acquire: drop stale cached h lines
    }
  }
}

// ---------------------------------------------------------------------------
extern "C" void kernel_launch(void* const* d_in, const int* in_sizes, int n_in,
                              void* d_out, int out_size, void* d_ws, size_t ws_size,
                              hipStream_t stream) {
  const float* X     = (const float*)d_in[0];   // (32,256,512)
  const float* Winit = (const float*)d_in[1];   // (1024,512)
  const float* Wih   = (const float*)d_in[2];   // (4,2,2048,1024)
  const float* Whh   = (const float*)d_in[3];   // (4,2,2048,512)
  const float* bih   = (const float*)d_in[4];   // (4,2,2048)
  const float* bhh   = (const float*)d_in[5];   // (4,2,2048)
  float* outp = (float*)d_out;                  // (32,256,1024)

  char* ws = (char*)d_ws;
  size_t off = 0;
  auto alloc = [&](size_t bytes) -> void* {
    off = (off + 255) & ~(size_t)255;
    void* p = ws + off;
    off += bytes;
    return p;
  };

  // recurrence sync state (zeroed every call): counters + h double buffers
  unsigned int* ctr = (unsigned int*)alloc(256);
  __bf16* hglob = (__bf16*)alloc((size_t)2 * 2 * BB * HH * 2);   // 128 KB
  size_t sync_bytes = off;   // contiguous from ws start

  // packed fragment buffers: (rows/16)*(K/32)*1024 bytes each
  __bf16* Xp     = (__bf16*)alloc((size_t)(MROWS/16) * (DIN/32) * 1024);  // 8 MB
  __bf16* Winitp = (__bf16*)alloc((size_t)(DD/16)    * (DIN/32) * 1024);  // 1 MB
  __bf16* Wihp[NLAYER];
  __bf16* Whhp[NLAYER];
  for (int L = 0; L < NLAYER; ++L) {
    Wihp[L] = (__bf16*)alloc((size_t)(NC/16) * (DD/32) * 1024);           // 8 MB
    Whhp[L] = (__bf16*)alloc((size_t)(NC/16) * (HH/32) * 1024);           // 4 MB
  }
  float* ybuf0 = (float*)alloc((size_t)MROWS * DD * 4);                   // 32 MB
  float* ybuf1 = (float*)alloc((size_t)MROWS * DD * 4);                   // 32 MB
  __bf16* Ypack = (__bf16*)alloc((size_t)(MROWS/16) * (DD/32) * 1024);    // 16 MB
  float* XG = (float*)alloc((size_t)MROWS * NC * 4);                      // 128 MB

  // ---- pack inputs & weights (bf16, WMMA fragment layouts) ----
  {
    int tot = (MROWS/16) * (DIN/32) * 32;
    pack_a_bf16<<<(tot + 255) / 256, 256, 0, stream>>>(X, Xp, MROWS/16, DIN/32, DIN);
  }
  {
    int tot = (DD/16) * (DIN/32) * 32;
    pack_b_bf16<<<(tot + 255) / 256, 256, 0, stream>>>(Winit, Winitp, DD/16, DIN/32, DIN);
  }
  for (int L = 0; L < NLAYER; ++L) {
    int tot1 = (NC/16) * (DD/32) * 32;
    pack_b_bf16<<<(tot1 + 255) / 256, 256, 0, stream>>>(
        Wih + (size_t)L * NC * DD, Wihp[L], NC/16, DD/32, DD);
    int tot2 = (NC/16) * (HH/32) * 32;
    pack_b_bf16<<<(tot2 + 255) / 256, 256, 0, stream>>>(
        Whh + (size_t)L * NC * HH, Whhp[L], NC/16, HH/32, HH);
  }

  // ---- initial bias-free linear: Y0 = X @ Winit^T ----
  gemm_wmma_bf16<<<dim3(DD / 128, MROWS / 128), 256, 0, stream>>>(
      (const v16bf*)Xp, (const v16bf*)Winitp, ybuf0, DD, DIN / 32,
      nullptr, nullptr);

  // ---- LSTM stack ----
  float* ping = ybuf0;
  float* pong = ybuf1;
  for (int L = 0; L < NLAYER; ++L) {
    // repack current activations as WMMA A fragments
    int tot = (MROWS/16) * (DD/32) * 32;
    pack_a_bf16<<<(tot + 255) / 256, 256, 0, stream>>>(ping, Ypack, MROWS/16, DD/32, DD);

    // x-gate precompute for both directions, biases folded in
    gemm_wmma_bf16<<<dim3(NC / 128, MROWS / 128), 256, 0, stream>>>(
        (const v16bf*)Ypack, (const v16bf*)Wihp[L], XG, NC, DD / 32,
        bih + (size_t)L * NC, bhh + (size_t)L * NC);

    // zero barrier counters + initial h buffers, then distributed recurrence
    hipMemsetAsync(d_ws, 0, sync_bytes, stream);
    float* dst = (L == NLAYER - 1) ? outp : pong;
    lstm_layer_recur<<<2 * NWGD, 128, 0, stream>>>(
        (const v16bf*)Whhp[L], XG, ping, dst, hglob, ctr);

    float* tmp = ping; ping = pong; pong = tmp;
  }
}
